// _SelfAttention_78589311582281
// MI455X (gfx1250) — matrix operand order repaired and checked by reading, not compiled, not run on hardware
//
#include <hip/hip_runtime.h>
#include <math.h>

typedef __attribute__((ext_vector_type(16))) _Float16 v16h;
typedef __attribute__((ext_vector_type(4)))  _Float16 v4h;
typedef __attribute__((ext_vector_type(8)))  float    v8f;
typedef __attribute__((ext_vector_type(4)))  float    v4f;
typedef int b128 __attribute__((vector_size(16)));   // 16-byte payload type for async LDS DMA

#define S_LEN  2048
#define DMODEL 2048
#define NHEADS 32
#define HD     64
#define BATCH  2
#define LDT    72   // padded LDS row stride (f16 elements) for 64-wide tiles

#if defined(__gfx1250__) && __has_builtin(__builtin_amdgcn_global_load_async_to_lds_b128)
#define ASYNC_LDS 1
#else
#define ASYNC_LDS 0
#endif

// 16-byte global -> LDS copy: async DMA (gfx1250 GLOBAL_LOAD_ASYNC_TO_LDS_B128) or via regs
__device__ __forceinline__ void cp16(const _Float16* g, _Float16* l) {
#if ASYNC_LDS
  __builtin_amdgcn_global_load_async_to_lds_b128(
      (__attribute__((address_space(1))) b128*)g,
      (__attribute__((address_space(3))) b128*)l, 0, 0);
#else
  *(uint4*)l = *(const uint4*)g;
#endif
}
__device__ __forceinline__ void async_wait() {
#if ASYNC_LDS
#if __has_builtin(__builtin_amdgcn_s_wait_asynccnt)
  __builtin_amdgcn_s_wait_asynccnt(0);
#else
  asm volatile("s_wait_asynccnt 0x0" ::: "memory");
#endif
#endif
}

union Frag { v16h v; uint4 q[2]; };

// A fragment (16x32 f16, M x K): lane (m=ln, half g) holds K={g*8..g*8+7, 16+g*8..16+g*8+7}
__device__ __forceinline__ v16h frag_A(const _Float16* base, int ld, int g, int ln) {
  Frag f;
  const _Float16* r = base + ln * ld + g * 8;
  f.q[0] = *(const uint4*)(r);
  f.q[1] = *(const uint4*)(r + 16);
  return f.v;
}
// B fragment (32x16 f16, K x N) from rows of B^T (row n, inner k): lane holds K={g*8..g*8+7, 16+g*8..16+g*8+7}
__device__ __forceinline__ v16h frag_B(const _Float16* base, int ld, int g, int ln) {
  Frag f;
  const _Float16* r = base + ln * ld + g * 8;
  f.q[0] = *(const uint4*)(r);
  f.q[1] = *(const uint4*)(r + 16);
  return f.v;
}
__device__ __forceinline__ v8f wmma_f16(v16h a, v16h b, v8f c) {
  return __builtin_amdgcn_wmma_f32_16x16x32_f16(false, a, false, b, (short)0, c, false, false);
}

// ---------------------------------------------------------------- cast f32 -> f16
__global__ __launch_bounds__(256) void cast_f32_f16(const float* __restrict__ s,
                                                    _Float16* __restrict__ d, int n) {
  int idx = (blockIdx.x * 256 + threadIdx.x) * 4;
  if (idx < n) {
    float4 f = *(const float4*)(s + idx);
    v4h o = { (_Float16)f.x, (_Float16)f.y, (_Float16)f.z, (_Float16)f.w };
    *(v4h*)(d + idx) = o;
  }
}

// ------------------------------------------- bias LUT: tab[h][dist+2047], dist in [-2047,2047]
__global__ __launch_bounds__(256) void bias_table(const float* __restrict__ rel,
                                                  float* __restrict__ tab) {
  int i = blockIdx.x * 256 + threadIdx.x;
  if (i >= NHEADS * 4095) return;
  int h    = i / 4095;
  int dist = i % 4095 - 2047;
  int rel0 = (dist > 0) ? 16 : 0;             // max_bucket = 16
  int ad   = dist < 0 ? -dist : dist;
  int bkt;
  if (ad < 8) {                               // limit = 8
    bkt = ad;
  } else {
    // 8 + int( log(ad/8) / log(128/8) * 8 ), clamped to 15
    int lg = 8 + (int)(logf((float)ad * 0.125f) * (8.0f / logf(16.0f)));
    bkt = lg < 15 ? lg : 15;
  }
  tab[i] = rel[(rel0 + bkt) * NHEADS + h];
}

// -------------------------------------- write position_bias [H,S,S] f32 (512 MB) from LUT
__global__ __launch_bounds__(256) void bias_out(const float* __restrict__ tab,
                                                float* __restrict__ out) {
  const size_t total4 = (size_t)NHEADS * S_LEN * S_LEN / 4;
  size_t stride = (size_t)gridDim.x * 256;
  for (size_t i4 = (size_t)blockIdx.x * 256 + threadIdx.x; i4 < total4; i4 += stride) {
    size_t i = i4 * 4;
    int k = (int)(i & 2047);
    int q = (int)((i >> 11) & 2047);
    int h = (int)(i >> 22);
    const float* bt = tab + (size_t)h * 4095 + 2047 - q + k;
    v4f o = { bt[0], bt[1], bt[2], bt[3] };
    __builtin_nontemporal_store(o, (v4f*)(out + i));   // NT: don't thrash L2
  }
}

// -------------------------------------------------------- 128x128-tile WMMA GEMM
// C[M x 2048] = A[M x 2048] * W[2048 x 2048]^T   (both row-major, K inner)
// Double-buffered LDS: staging never crosses a barrier -> no cross-barrier spills.
// mode 0: f16 scatter into [B,H,S,HD] (QKV)   mode 1: f32 into [M, 2048] (out proj)
__global__ __launch_bounds__(256) void gemm128(const _Float16* __restrict__ A,
                                               const _Float16* __restrict__ W,
                                               void* __restrict__ dst, int mode) {
  __shared__ alignas(16) _Float16 As[2][128 * LDT];
  __shared__ alignas(16) _Float16 Ws[2][128 * LDT];
  const int K  = DMODEL;
  const int m0 = blockIdx.y * 128;
  const int n0 = blockIdx.x * 128;
  const int t  = threadIdx.x;
  const int w  = t >> 5, l = t & 31;
  const int wm = (w & 3) * 32, wn = (w >> 2) * 64;
  const int g  = l >> 4, ln = l & 15;
  const int lrow = t >> 3;              // staging: 1024 x 16B per operand, 4 per thread
  const int c8   = (t & 7) * 8;

  auto stage = [&](int k0, int buf) {
#pragma unroll
    for (int i = 0; i < 4; ++i) {
      int r = lrow + 32 * i;
      cp16(A + (size_t)(m0 + r) * K + k0 + c8, &As[buf][r * LDT + c8]);
      cp16(W + (size_t)(n0 + r) * K + k0 + c8, &Ws[buf][r * LDT + c8]);
    }
  };

  v8f acc[2][4] = {};
  stage(0, 0);
  async_wait();
  __syncthreads();
  for (int k0 = 0; k0 < K; k0 += 64) {
    const int cur = (k0 >> 6) & 1;
    if (k0 + 64 < K) stage(k0 + 64, cur ^ 1);   // fill other buffer, no barrier needed
    const _Float16* Ab = As[cur];
    const _Float16* Wb = Ws[cur];
#pragma unroll
    for (int kk = 0; kk < 64; kk += 32) {
      v16h af0 = frag_A(Ab + (wm     ) * LDT + kk, LDT, g, ln);
      v16h af1 = frag_A(Ab + (wm + 16) * LDT + kk, LDT, g, ln);
      v16h bf[4];
#pragma unroll
      for (int tn = 0; tn < 4; ++tn)
        bf[tn] = frag_B(Wb + (wn + tn * 16) * LDT + kk, LDT, g, ln);
#pragma unroll
      for (int tn = 0; tn < 4; ++tn) {
        acc[0][tn] = wmma_f16(af0, bf[tn], acc[0][tn]);
        acc[1][tn] = wmma_f16(af1, bf[tn], acc[1][tn]);
      }
    }
    async_wait();
    __syncthreads();                            // next tile landed; cur reads done
  }
  if (mode == 0) {
    _Float16* o = (_Float16*)dst;       // [B*H, S, HD]
#pragma unroll
    for (int tm = 0; tm < 2; ++tm)
#pragma unroll
      for (int tn = 0; tn < 4; ++tn)
#pragma unroll
        for (int v = 0; v < 8; ++v) {
          int m = m0 + wm + tm * 16 + 8 * g + v;
          int n = n0 + wn + tn * 16 + ln;
          int b = m >> 11, s = m & 2047;
          int h = n >> 6,  d = n & 63;
          o[(((size_t)b * NHEADS + h) * S_LEN + s) * HD + d] = (_Float16)acc[tm][tn][v];
        }
  } else {
    float* o = (float*)dst;             // [M, 2048] f32
#pragma unroll
    for (int tm = 0; tm < 2; ++tm)
#pragma unroll
      for (int tn = 0; tn < 4; ++tn)
#pragma unroll
        for (int v = 0; v < 8; ++v) {
          int m = m0 + wm + tm * 16 + 8 * g + v;
          int n = n0 + wn + tn * 16 + ln;
          o[(size_t)m * DMODEL + n] = acc[tm][tn][v];
        }
  }
}

// ------------------------------------------------ flash attention, 64 queries / block
// grid (S/64, B*H); 128 threads = 4 waves; each wave owns 16 query rows x full hd.
// K/V tiles double-buffered in LDS; P re-layout (C->A fragment) through per-wave LDS.
__global__ __launch_bounds__(128) void flash_attn(const _Float16* __restrict__ Q,
                                                  const _Float16* __restrict__ Km,
                                                  const _Float16* __restrict__ Vm,
                                                  const float* __restrict__ btab,
                                                  _Float16* __restrict__ ctx) {
  __shared__ alignas(16) _Float16 Ks [2][64 * LDT];    // K tile  [key][d]
  __shared__ alignas(16) _Float16 Vts[2][64 * LDT];    // V tile transposed [d][key]
  __shared__ alignas(16) _Float16 Ps [4][16 * LDT];    // per-wave P buffer
  const int q0 = blockIdx.x * 64;
  const int bh = blockIdx.y;
  const int h  = bh & (NHEADS - 1);
  const int t  = threadIdx.x, w = t >> 5, l = t & 31, g = l >> 4, ln = l & 15;
  const _Float16* Qb = Q  + (size_t)bh * S_LEN * HD;
  const _Float16* Kb = Km + (size_t)bh * S_LEN * HD;
  const _Float16* Vb = Vm + (size_t)bh * S_LEN * HD;
  const int lrow = t >> 3;
  const int c8   = (t & 7) * 8;
  const int qrow = q0 + w * 16;

  auto stageKV = [&](int kb, int buf) {
#pragma unroll
    for (int i = 0; i < 4; ++i) {
      int r = lrow + 16 * i;
      cp16(Kb + (size_t)(kb + r) * HD + c8, &Ks[buf][r * LDT + c8]);
      union { uint4 q; _Float16 e[8]; } u;             // V transposed through regs
      u.q = *(const uint4*)(Vb + (size_t)(kb + r) * HD + c8);
#pragma unroll
      for (int e = 0; e < 8; ++e) Vts[buf][(c8 + e) * LDT + r] = u.e[e];
    }
  };

  v16h aq0 = frag_A(Qb + (size_t)qrow * HD,      HD, g, ln);   // K(d) = 0..31
  v16h aq1 = frag_A(Qb + (size_t)qrow * HD + 32, HD, g, ln);   // K(d) = 32..63

  v8f acc[4] = {};
  float mrun[8], lrun[8];
#pragma unroll
  for (int v = 0; v < 8; ++v) { mrun[v] = -3.0e38f; lrun[v] = 0.f; }
  const float* bt = btab + (size_t)h * 4095 + 2047 - qrow - 8 * g;

  stageKV(0, 0);
  async_wait();
  __syncthreads();
  for (int kb = 0; kb < S_LEN; kb += 64) {
    const int cur = (kb >> 6) & 1;
    if (kb + 64 < S_LEN) stageKV(kb + 64, cur ^ 1);    // prefetch next tile into buf^1

    // scores S[16 x 64] = Q * K^T  (+ relative-position bias)
    const _Float16* Kc = Ks[cur];
    v8f st[4];
#pragma unroll
    for (int tn = 0; tn < 4; ++tn) {
      v8f s = {};
      v16h b0 = frag_B(Kc + (tn * 16) * LDT,      LDT, g, ln);   // d = 0..31
      v16h b1 = frag_B(Kc + (tn * 16) * LDT + 32, LDT, g, ln);   // d = 32..63
      s = wmma_f16(aq0, b0, s);
      s = wmma_f16(aq1, b1, s);
      const int key = kb + tn * 16 + ln;
#pragma unroll
      for (int v = 0; v < 8; ++v) s[v] += bt[key - v];           // tab[h][key-q+2047]
      st[tn] = s;
    }

    // online softmax per query row (row m = 8g+v; lanes with equal g share rows)
#pragma unroll
    for (int v = 0; v < 8; ++v) {
      float mx = fmaxf(fmaxf(st[0][v], st[1][v]), fmaxf(st[2][v], st[3][v]));
#pragma unroll
      for (int off = 1; off < 16; off <<= 1) mx = fmaxf(mx, __shfl_xor(mx, off, 32));
      float mnew = fmaxf(mrun[v], mx);
      float corr = __expf(mrun[v] - mnew);
      mrun[v] = mnew;
      float rs = 0.f;
#pragma unroll
      for (int tn = 0; tn < 4; ++tn) {
        float p = __expf(st[tn][v] - mnew);
        st[tn][v] = p;
        rs += p;
      }
#pragma unroll
      for (int off = 1; off < 16; off <<= 1) rs += __shfl_xor(rs, off, 32);
      lrun[v] = lrun[v] * corr + rs;
#pragma unroll
      for (int td = 0; td < 4; ++td) acc[td][v] *= corr;
    }

    // re-layout P (C layout -> A layout) through per-wave LDS
    _Float16* Pw = Ps[w];
#pragma unroll
    for (int tn = 0; tn < 4; ++tn)
#pragma unroll
      for (int v = 0; v < 8; ++v)
        Pw[(8 * g + v) * LDT + tn * 16 + ln] = (_Float16)st[tn][v];
    __syncthreads();                   // Ps visible (buf^1 fills continue in flight)

    // ctx += P[16 x 64] * V[64 x 64]
    const _Float16* Vc = Vts[cur];
    v16h ap0 = frag_A(Pw,      LDT, g, ln);                      // key = 0..31
    v16h ap1 = frag_A(Pw + 32, LDT, g, ln);                      // key = 32..63
#pragma unroll
    for (int td = 0; td < 4; ++td) {
      v16h vb0 = frag_B(Vc + (td * 16) * LDT,      LDT, g, ln);
      v16h vb1 = frag_B(Vc + (td * 16) * LDT + 32, LDT, g, ln);
      acc[td] = wmma_f16(ap0, vb0, acc[td]);
      acc[td] = wmma_f16(ap1, vb1, acc[td]);
    }
    async_wait();
    __syncthreads();                   // buf^1 complete; all reads of cur done
  }

  const int b = bh >> 5;
#pragma unroll
  for (int td = 0; td < 4; ++td)
#pragma unroll
    for (int v = 0; v < 8; ++v) {
      int srow = qrow + 8 * g + v;
      float o = acc[td][v] / lrun[v];
      ctx[((size_t)b * S_LEN + srow) * DMODEL + h * HD + td * 16 + ln] = (_Float16)o;
    }
}

// ---------------------------------------------------------------------------- launch
extern "C" void kernel_launch(void* const* d_in, const int* in_sizes, int n_in,
                              void* d_out, int out_size, void* d_ws, size_t ws_size,
                              hipStream_t stream) {
  (void)in_sizes; (void)n_in; (void)out_size; (void)ws_size;
  const float* x  = (const float*)d_in[0];
  const float* wq = (const float*)d_in[1];
  const float* wk = (const float*)d_in[2];
  const float* wv = (const float*)d_in[3];
  const float* wo = (const float*)d_in[4];
  const float* rb = (const float*)d_in[5];

  const size_t NX = (size_t)BATCH * S_LEN * DMODEL;   // 8388608
  const size_t NW = (size_t)DMODEL * DMODEL;          // 4194304
  _Float16* xh  = (_Float16*)d_ws;                    // ~118 MB workspace total
  _Float16* wqh = xh  + NX;
  _Float16* wkh = wqh + NW;
  _Float16* wvh = wkh + NW;
  _Float16* woh = wvh + NW;
  _Float16* Qh  = woh + NW;                           // [B,H,S,HD] f16
  _Float16* Kh  = Qh  + NX;
  _Float16* Vh  = Kh  + NX;
  _Float16* ch  = Vh  + NX;                           // ctx [B*S, D] f16
  float*   btab = (float*)(ch + NX);                  // [H, 4095] f32 LUT

  cast_f32_f16<<<dim3((unsigned)((NX / 4 + 255) / 256)), dim3(256), 0, stream>>>(x,  xh,  (int)NX);
  cast_f32_f16<<<dim3((unsigned)((NW / 4 + 255) / 256)), dim3(256), 0, stream>>>(wq, wqh, (int)NW);
  cast_f32_f16<<<dim3((unsigned)((NW / 4 + 255) / 256)), dim3(256), 0, stream>>>(wk, wkh, (int)NW);
  cast_f32_f16<<<dim3((unsigned)((NW / 4 + 255) / 256)), dim3(256), 0, stream>>>(wv, wvh, (int)NW);
  cast_f32_f16<<<dim3((unsigned)((NW / 4 + 255) / 256)), dim3(256), 0, stream>>>(wo, woh, (int)NW);

  bias_table<<<dim3((NHEADS * 4095 + 255) / 256), dim3(256), 0, stream>>>(rb, btab);

  float* out      = (float*)d_out;
  float* bias_dst = out + NX;                         // tuple: (out, position_bias)
  bias_out<<<dim3(8192), dim3(256), 0, stream>>>(btab, bias_dst);

  dim3 gg(DMODEL / 128, (unsigned)((BATCH * S_LEN) / 128));   // (16, 32)
  gemm128<<<gg, dim3(256), 0, stream>>>(xh, wqh, Qh, 0);
  gemm128<<<gg, dim3(256), 0, stream>>>(xh, wkh, Kh, 0);
  gemm128<<<gg, dim3(256), 0, stream>>>(xh, wvh, Vh, 0);

  flash_attn<<<dim3(S_LEN / 64, BATCH * NHEADS), dim3(128), 0, stream>>>(Qh, Kh, Vh, btab, ch);

  gemm128<<<gg, dim3(256), 0, stream>>>(ch, woh, (void*)out, 1);
}
